// Decoder_Flow_86663850099204
// MI455X (gfx1250) — matrix operand order repaired and checked by reading, not compiled, not run on hardware
//
#include <hip/hip_runtime.h>
#include <stdint.h>

// ---------------------------------------------------------------------------
// CNF (ConcatSquash ODE net, RK4 x10, Hutchinson divergence) for MI455X.
// f16 WMMA for the 256x256 layers (fwd + VJP fused), weights staged into LDS
// per-GEMM via the Tensor Data Mover (TENSOR_LOAD_TO_LDS + s_wait_tensorcnt),
// so B fragments are ds_load_b128 (no register-file blowout / scratch spills).
// ---------------------------------------------------------------------------

#define USE_TDM 1

typedef __attribute__((ext_vector_type(16))) _Float16 v16h;
typedef __attribute__((ext_vector_type(8)))  float    v8f;
typedef __attribute__((ext_vector_type(4)))  unsigned int u32x4;
typedef __attribute__((ext_vector_type(8)))  int      i32x8;
typedef __attribute__((ext_vector_type(4)))  int      i32x4;

#define NCTX    32
#define KPTS    1024
#define ZDIM    128
#define NSTEPS  10
#define NTVALS  21        // distinct t values: 0, 0.05, ..., 1.0
#define TM      64        // rows (points) per block
#define BSTRIDE 264       // LDS act row stride in halves (256 + 8 pad)
#define DTV     0.1f
#define HDTV    0.05f

union FragH { v16h v; uint32_t u[8]; _Float16 h[16]; };
union PairH { uint32_t u; _Float16 h[2]; };

// ---------------------------------------------------------------------------
// Pack 256x256 f32 weight into WMMA B-fragment order (f16), XOR-swizzled so
// the per-lane 2x ds_load_b128 in the GEMM are bank-friendly.
// B (32x16, KxN): lane<16 -> N=lane,    half h -> K = h
//                 lane>=16-> N=lane-16, half h -> K = 16+h
// Per tile (kt,nt): 256 dwords; dword for (lane,j) lives at
//   chunk = (lane*2 + (j>>2)) ^ ((lane>>3)&1);  index = chunk*4 + (j&3)
// transpose=1 packs W^T (K-dim = out-dim of W).
// ---------------------------------------------------------------------------
__global__ void pack_w256(const float* __restrict__ W, _Float16* __restrict__ frag,
                          int transpose)
{
    int gid  = blockIdx.x * blockDim.x + threadIdx.x;   // 0..4095
    int tile = gid >> 5;                                // 0..127  (8 kt x 16 nt)
    int lane = gid & 31;
    int nt = tile & 15;
    int kt = tile >> 4;
    int nn = (nt << 4) + (lane & 15);
    int kbase = (kt << 5) + ((lane < 16) ? 0 : 16);
    int sw = (lane >> 3) & 1;
    uint32_t* dstu = (uint32_t*)frag + (size_t)tile * 256;
    #pragma unroll
    for (int j = 0; j < 8; ++j) {
        int k0 = kbase + 2 * j, k1 = k0 + 1;
        float e0 = transpose ? W[nn * 256 + k0] : W[k0 * 256 + nn];
        float e1 = transpose ? W[nn * 256 + k1] : W[k1 * 256 + nn];
        PairH p; p.h[0] = (_Float16)e0; p.h[1] = (_Float16)e1;
        int chunk = (lane * 2 + (j >> 2)) ^ sw;
        dstu[chunk * 4 + (j & 3)] = p.u;
    }
}

// ---------------------------------------------------------------------------
// Precompute gate = sigmoid([t,h] @ Wg + bg) and bias = [t,h] @ Wb for every
// (t-index, layer, context). Table layout: [(j*4+layer)*32 + n][256].
// ---------------------------------------------------------------------------
__global__ void gate_kernel(const float* __restrict__ h,
                            const float* __restrict__ Wg,
                            const float* __restrict__ bg,
                            const float* __restrict__ Wb,
                            float* __restrict__ gGate,
                            float* __restrict__ gBias,
                            int layer, int dl)
{
    int b = blockIdx.x;          // 0 .. NTVALS*32-1
    int j = b >> 5;
    int n = b & 31;
    int c = threadIdx.x;
    float t = (float)j * HDTV;
    float g = 0.f, bb = 0.f;
    if (c < dl) {
        g  = t * Wg[c];
        bb = t * Wb[c];
        const float* hn = h + (size_t)n * ZDIM;
        for (int i = 0; i < ZDIM; ++i) {
            float hv = hn[i];
            g  += hv * Wg[(size_t)(1 + i) * dl + c];
            bb += hv * Wb[(size_t)(1 + i) * dl + c];
        }
        g += bg[c];
        g = 1.0f / (1.0f + expf(-g));
    }
    size_t o = ((size_t)(j * 4 + layer) * 32 + n) * 256 + c;
    gGate[o] = (c < dl) ? g  : 0.f;
    gBias[o] = (c < dl) ? bb : 0.f;
}

// ---------------------------------------------------------------------------
// TDM: stage one 128KB packed weight buffer into LDS (issued by one wave).
// 1-D descriptor: data_size=8B, tile_dim0 = tensor_dim0 = 16384 elements.
// ---------------------------------------------------------------------------
#if USE_TDM
__device__ __forceinline__ void tdm_load_weights(const _Float16* gsrc,
                                                 _Float16* ldst)
{
    uint64_t ga = (uint64_t)(uintptr_t)gsrc;
    uint32_t la = (uint32_t)(uintptr_t)(void*)ldst;   // LDS byte offset
    u32x4 g0;
    g0[0] = 1u;                                        // count=1, user desc
    g0[1] = la;                                        // lds_addr
    g0[2] = (uint32_t)ga;                              // global_addr[31:0]
    g0[3] = (uint32_t)((ga >> 32) & 0x01FFFFFFu) | (2u << 30); // type=2
    i32x8 g1;
    g1[0] = (int)(3u << 16);        // workgroup_mask=0, data_size=8B
    g1[1] = (int)(16384u << 16);    // tensor_dim0[15:0] at bits 48..63
    g1[2] = (int)(1u << 16);        // tensor_dim0 hi=0 | tensor_dim1=1
    g1[3] = (int)(16384u << 16);    // tile_dim0 = 16384
    g1[4] = 0;                      // tile_dim1/2 unused
    g1[5] = 16384;                  // tensor_dim0_stride
    g1[6] = 0;
    g1[7] = 0;
    i32x4 z4 = {0, 0, 0, 0};
    i32x8 z8 = {0, 0, 0, 0, 0, 0, 0, 0};
    __builtin_amdgcn_tensor_load_to_lds(g0, g1, z4, z4, z8, 0);
}
#define STAGE_ISSUE(SRC)                                        \
    do { if (wave == 0) tdm_load_weights((SRC), wbuf); } while (0)
#define STAGE_WAIT()                                            \
    do { if (wave == 0) __builtin_amdgcn_s_wait_tensorcnt(0); } while (0)
#else
__device__ __forceinline__ void stage_weights_sync(const _Float16* __restrict__ g,
                                                   _Float16* l, int tid)
{
    const uint4* s = (const uint4*)g;
    uint4* d = (uint4*)l;
    for (int i = tid; i < 8192; i += 256) d[i] = s[i];
}
#define STAGE_ISSUE(SRC) stage_weights_sync((SRC), wbuf, tid)
#define STAGE_WAIT()     do { } while (0)
#endif

// ---------------------------------------------------------------------------
// One 64x256x256 GEMM: acc = A(LDS f16, row-major, stride BSTRIDE) x B(LDS frag).
// 8 waves: wave w -> M-tile (w>>1), N-tiles (w&1)*8 .. +7.
// A (16x32): lane<16 -> M=lane,    halves 0..7 K=0..7,  8..15 K=8..15
//            lane>=16-> M=lane-16, halves 0..7 K=16..23, 8..15 K=24..31
// ---------------------------------------------------------------------------
__device__ __forceinline__ void gemm64(const _Float16* bufIn,
                                       const _Float16* wbufL,
                                       int wave, int lane, v8f acc[8])
{
    const int mt  = wave >> 1;
    const int ntb = (wave & 1) * 8;
    const v8f zero = {0.f, 0.f, 0.f, 0.f, 0.f, 0.f, 0.f, 0.f};
    #pragma unroll
    for (int i = 0; i < 8; ++i) acc[i] = zero;

    const int m  = mt * 16 + (lane & 15);
    const int b2 = (lane < 16) ? 0 : 8;          // A uint-index offset
    const int sw = (lane >> 3) & 1;
    const int cA = ((lane * 2) ^ sw) * 4;        // B chunk bases (swizzled)
    const int cB = ((lane * 2 + 1) ^ sw) * 4;

    for (int kt = 0; kt < 8; ++kt) {
        FragH fa;
        const uint32_t* r32 = (const uint32_t*)(bufIn + m * BSTRIDE + kt * 32);
        #pragma unroll
        for (int jj = 0; jj < 4; ++jj) fa.u[jj]     = r32[b2 + jj];
        #pragma unroll
        for (int jj = 0; jj < 4; ++jj) fa.u[4 + jj] = r32[4 + b2 + jj];

        #pragma unroll
        for (int i = 0; i < 8; ++i) {
            FragH fb;
            const uint32_t* tb =
                (const uint32_t*)wbufL + (size_t)(kt * 16 + (ntb + i)) * 256;
            #pragma unroll
            for (int jj = 0; jj < 4; ++jj) fb.u[jj]     = tb[cA + jj];
            #pragma unroll
            for (int jj = 0; jj < 4; ++jj) fb.u[4 + jj] = tb[cB + jj];
            acc[i] = __builtin_amdgcn_wmma_f32_16x16x32_f16(
                false, fa.v, false, fb.v, (short)0, acc[i], false, false);
        }
    }
}

// ---------------------------------------------------------------------------
// Fused CNF integrator: each block integrates 64 points of one context.
// ---------------------------------------------------------------------------
__global__ __launch_bounds__(256)
void cnf_kernel(const float* __restrict__ x,  const float* __restrict__ ebuf,
                const float* __restrict__ W0, const float* __restrict__ b0,
                const float* __restrict__ b1, const float* __restrict__ b2,
                const float* __restrict__ W3, const float* __restrict__ b3,
                const _Float16* __restrict__ W1B, const _Float16* __restrict__ W2B,
                const _Float16* __restrict__ W1T, const _Float16* __restrict__ W2T,
                const float* __restrict__ gGate, const float* __restrict__ gBias,
                float* __restrict__ out)
{
    __shared__ __attribute__((aligned(16))) _Float16 wbuf[65536];   // 128KB staged weights
    __shared__ _Float16 buf1[TM * BSTRIDE];   // a1 / later dz0*g0
    __shared__ _Float16 buf2[TM * BSTRIDE];   // a2 / later dz1*g1
    __shared__ _Float16 buf3[TM * BSTRIDE];   // a3 / later dz2*g2
    __shared__ float sW0[3 * 256];            // [din=3][256]
    __shared__ float sW3[256 * 3];            // [256][dout=3]
    __shared__ float sb0[256], sb1[256], sb2[256], sb3[4];
    __shared__ float sG[4][256], sBi[4][256];
    __shared__ float ybase[TM * 3], ys[TM * 3], dyv[TM * 3], accY[TM * 3];
    __shared__ float tmp3[TM * 3], sE[TM * 3];
    __shared__ float accD[TM], kdv[TM], lp[TM];

    const int tid  = threadIdx.x;
    const int wave = tid >> 5;
    const int lane = tid & 31;
    const int n    = blockIdx.x >> 4;                 // context (16 tiles/context)
    const int rowbase = blockIdx.x * TM;

    for (int i = tid; i < 768; i += 256) { sW0[i] = W0[i]; sW3[i] = W3[i]; }
    if (tid < 256) { sb0[tid] = b0[tid]; sb1[tid] = b1[tid]; sb2[tid] = b2[tid]; }
    if (tid < 3)   sb3[tid] = b3[tid];
    for (int i = tid; i < TM * 3; i += 256) {
        float xv = x[(size_t)rowbase * 3 + i];
        ybase[i] = xv; ys[i] = xv;
        sE[i] = ebuf[(size_t)rowbase * 3 + i];
    }
    if (tid < TM) lp[tid] = 0.f;
    __syncthreads();

    for (int step = 0; step < NSTEPS; ++step) {
        for (int s = 0; s < 4; ++s) {
            const int tj = 2 * step + ((s == 0) ? 0 : (s == 3) ? 2 : 1);

            // ---- phase 1: TDM(W1) overlapped with gates copy -----------
            STAGE_ISSUE(W1B);
            {
                const size_t base = ((size_t)(tj * 4) * 32 + n) * 256;
                for (int i = tid; i < 4 * 256; i += 256) {
                    int l = i >> 8, c = i & 255;
                    size_t o = base + (size_t)l * 32 * 256 + c;
                    sG[l][c]  = gGate[o];
                    sBi[l][c] = gBias[o];
                }
            }
            STAGE_WAIT();
            __syncthreads();

            // ---- layer 0 (3->256, VALU) --------------------------------
            for (int idx = tid; idx < TM * 256; idx += 256) {
                int r = idx >> 8, c = idx & 255;
                float z = ys[r * 3]     * sW0[c]
                        + ys[r * 3 + 1] * sW0[256 + c]
                        + ys[r * 3 + 2] * sW0[512 + c] + sb0[c];
                z = z * sG[0][c] + sBi[0][c];
                buf1[r * BSTRIDE + c] = (_Float16)tanhf(z);
            }
            __syncthreads();

            // ---- layer 1 (WMMA): a2 = tanh((a1@W1+b1)*g1+bi1) ----------
            {
                v8f acc[8];
                gemm64(buf1, wbuf, wave, lane, acc);
                const int mt = wave >> 1, ntb = (wave & 1) * 8;
                #pragma unroll
                for (int i = 0; i < 8; ++i) {
                    #pragma unroll
                    for (int v = 0; v < 8; ++v) {
                        int r = mt * 16 + ((lane < 16) ? v : v + 8);
                        int c = (ntb + i) * 16 + (lane & 15);
                        float z = (acc[i][v] + sb1[c]) * sG[1][c] + sBi[1][c];
                        buf2[r * BSTRIDE + c] = (_Float16)tanhf(z);
                    }
                }
            }
            __syncthreads();

            // ---- stage W2 ----------------------------------------------
            STAGE_ISSUE(W2B);
            STAGE_WAIT();
            __syncthreads();

            // ---- layer 2 (WMMA): a3 ------------------------------------
            {
                v8f acc[8];
                gemm64(buf2, wbuf, wave, lane, acc);
                const int mt = wave >> 1, ntb = (wave & 1) * 8;
                #pragma unroll
                for (int i = 0; i < 8; ++i) {
                    #pragma unroll
                    for (int v = 0; v < 8; ++v) {
                        int r = mt * 16 + ((lane < 16) ? v : v + 8);
                        int c = (ntb + i) * 16 + (lane & 15);
                        float z = (acc[i][v] + sb2[c]) * sG[2][c] + sBi[2][c];
                        buf3[r * BSTRIDE + c] = (_Float16)tanhf(z);
                    }
                }
            }
            __syncthreads();

            // ---- layer 3 (256->3) -> dy, TDM(W2^T) overlapped ----------
            STAGE_ISSUE(W2T);
            for (int idx = tid; idx < TM * 3; idx += 256) {
                int r = idx / 3, d = idx - r * 3;
                const uint32_t* row = (const uint32_t*)(buf3 + r * BSTRIDE);
                float s0 = 0.f;
                for (int cc = 0; cc < 128; ++cc) {
                    PairH p; p.u = row[cc];
                    s0 += (float)p.h[0] * sW3[(2 * cc) * 3 + d];
                    s0 += (float)p.h[1] * sW3[(2 * cc + 1) * 3 + d];
                }
                dyv[idx] = (s0 + sb3[d]) * sG[3][d] + sBi[3][d];
            }
            STAGE_WAIT();
            __syncthreads();

            // ---- VJP seed: buf3 <- (e*g3 @ W3^T) * (1-a3^2) * g2 -------
            for (int idx = tid; idx < TM * 256; idx += 256) {
                int r = idx >> 8, c = idx & 255;
                float g3 = sE[r * 3]     * sG[3][0] * sW3[c * 3]
                         + sE[r * 3 + 1] * sG[3][1] * sW3[c * 3 + 1]
                         + sE[r * 3 + 2] * sG[3][2] * sW3[c * 3 + 2];
                float a3 = (float)buf3[r * BSTRIDE + c];
                buf3[r * BSTRIDE + c] = (_Float16)(g3 * (1.f - a3 * a3) * sG[2][c]);
            }
            __syncthreads();

            // ---- bwd layer 2 (WMMA, W2^T): buf2 <- ga2*(1-a2^2)*g1 -----
            {
                v8f acc[8];
                gemm64(buf3, wbuf, wave, lane, acc);
                const int mt = wave >> 1, ntb = (wave & 1) * 8;
                #pragma unroll
                for (int i = 0; i < 8; ++i) {
                    #pragma unroll
                    for (int v = 0; v < 8; ++v) {
                        int r = mt * 16 + ((lane < 16) ? v : v + 8);
                        int c = (ntb + i) * 16 + (lane & 15);
                        float a2 = (float)buf2[r * BSTRIDE + c];
                        buf2[r * BSTRIDE + c] =
                            (_Float16)(acc[i][v] * (1.f - a2 * a2) * sG[1][c]);
                    }
                }
            }
            __syncthreads();

            // ---- stage W1^T --------------------------------------------
            STAGE_ISSUE(W1T);
            STAGE_WAIT();
            __syncthreads();

            // ---- bwd layer 1 (WMMA, W1^T): buf1 <- ga1*(1-a1^2)*g0 -----
            {
                v8f acc[8];
                gemm64(buf2, wbuf, wave, lane, acc);
                const int mt = wave >> 1, ntb = (wave & 1) * 8;
                #pragma unroll
                for (int i = 0; i < 8; ++i) {
                    #pragma unroll
                    for (int v = 0; v < 8; ++v) {
                        int r = mt * 16 + ((lane < 16) ? v : v + 8);
                        int c = (ntb + i) * 16 + (lane & 15);
                        float a1 = (float)buf1[r * BSTRIDE + c];
                        buf1[r * BSTRIDE + c] =
                            (_Float16)(acc[i][v] * (1.f - a1 * a1) * sG[0][c]);
                    }
                }
            }
            __syncthreads();

            // ---- eJ = buf1 @ W0^T ; div = sum(eJ * e) ------------------
            for (int idx = tid; idx < TM * 3; idx += 256) {
                int r = idx / 3, d = idx - r * 3;
                const uint32_t* row = (const uint32_t*)(buf1 + r * BSTRIDE);
                const float* w0d = sW0 + d * 256;
                float s0 = 0.f;
                for (int cc = 0; cc < 128; ++cc) {
                    PairH p; p.u = row[cc];
                    s0 += (float)p.h[0] * w0d[2 * cc];
                    s0 += (float)p.h[1] * w0d[2 * cc + 1];
                }
                tmp3[idx] = s0 * sE[idx];
            }
            __syncthreads();
            if (tid < TM) kdv[tid] = tmp3[tid * 3] + tmp3[tid * 3 + 1] + tmp3[tid * 3 + 2];
            __syncthreads();

            // ---- RK4 stage combine -------------------------------------
            if (s == 0) {
                for (int i = tid; i < TM * 3; i += 256) {
                    accY[i] = dyv[i]; ys[i] = ybase[i] + 0.5f * DTV * dyv[i];
                }
                if (tid < TM) accD[tid] = kdv[tid];
            } else if (s == 1) {
                for (int i = tid; i < TM * 3; i += 256) {
                    accY[i] += 2.f * dyv[i]; ys[i] = ybase[i] + 0.5f * DTV * dyv[i];
                }
                if (tid < TM) accD[tid] += 2.f * kdv[tid];
            } else if (s == 2) {
                for (int i = tid; i < TM * 3; i += 256) {
                    accY[i] += 2.f * dyv[i]; ys[i] = ybase[i] + DTV * dyv[i];
                }
                if (tid < TM) accD[tid] += 2.f * kdv[tid];
            } else {
                for (int i = tid; i < TM * 3; i += 256) {
                    float yn = ybase[i] + (DTV / 6.f) * (accY[i] + dyv[i]);
                    ybase[i] = yn; ys[i] = yn;
                }
                if (tid < TM) lp[tid] -= (DTV / 6.f) * (accD[tid] + kdv[tid]);
            }
            __syncthreads();
        }
    }

    for (int i = tid; i < TM * 3; i += 256)
        out[(size_t)rowbase * 3 + i] = ybase[i];
    float* outlp = out + (size_t)NCTX * KPTS * 3;
    if (tid < TM) outlp[rowbase + tid] = lp[tid];
}

// ---------------------------------------------------------------------------
extern "C" void kernel_launch(void* const* d_in, const int* in_sizes, int n_in,
                              void* d_out, int out_size, void* d_ws, size_t ws_size,
                              hipStream_t stream)
{
    (void)in_sizes; (void)n_in; (void)out_size; (void)ws_size;

    const float* x = (const float*)d_in[0];
    const float* h = (const float*)d_in[1];
    const float* e = (const float*)d_in[2];
    const float *W[4], *bv[4], *Wg[4], *bgp[4], *Wb[4];
    for (int l = 0; l < 4; ++l) {
        W[l]   = (const float*)d_in[3 + 5 * l + 0];
        bv[l]  = (const float*)d_in[3 + 5 * l + 1];
        Wg[l]  = (const float*)d_in[3 + 5 * l + 2];
        bgp[l] = (const float*)d_in[3 + 5 * l + 3];
        Wb[l]  = (const float*)d_in[3 + 5 * l + 4];
    }

    char* ws = (char*)d_ws;
    const size_t gateElems = (size_t)NTVALS * 4 * 32 * 256;
    float* gGate = (float*)ws;  ws += gateElems * sizeof(float);
    float* gBias = (float*)ws;  ws += gateElems * sizeof(float);
    _Float16* W1B = (_Float16*)ws; ws += (size_t)65536 * 2;
    _Float16* W2B = (_Float16*)ws; ws += (size_t)65536 * 2;
    _Float16* W1T = (_Float16*)ws; ws += (size_t)65536 * 2;
    _Float16* W2T = (_Float16*)ws; ws += (size_t)65536 * 2;

    pack_w256<<<16, 256, 0, stream>>>(W[1], W1B, 0);
    pack_w256<<<16, 256, 0, stream>>>(W[2], W2B, 0);
    pack_w256<<<16, 256, 0, stream>>>(W[1], W1T, 1);
    pack_w256<<<16, 256, 0, stream>>>(W[2], W2T, 1);

    for (int l = 0; l < 4; ++l) {
        int dl = (l == 3) ? 3 : 256;
        gate_kernel<<<NTVALS * 32, 256, 0, stream>>>(h, Wg[l], bgp[l], Wb[l],
                                                     gGate, gBias, l, dl);
    }

    cnf_kernel<<<(NCTX * KPTS) / TM, 256, 0, stream>>>(
        x, e, W[0], bv[0], bv[1], bv[2], W[3], bv[3],
        W1B, W2B, W1T, W2T, gGate, gBias, (float*)d_out);
}
